// WLConvMultiFeature_25220047962707
// MI455X (gfx1250) — 8-bit matrix operand order repaired and checked by reading, compiled, not run on hardware
//
#include <hip/hip_runtime.h>
#include <hip/hip_bf16.h>


typedef __attribute__((ext_vector_type(8))) int v8i;

#define WL_C1 6364136223846793005LL
#define WL_C2 1442695040888963407LL
#define WL_C3 2862933555777941757LL
#define WL_C4 3202034522624059733LL

#define TAB_BITS 21
#define TAB_SIZE (1u << TAB_BITS)
#define KEY_EMPTY 0xFFFFFFFFFFFFFFFFull

#if defined(__HIP_DEVICE_COMPILE__) && defined(__has_builtin)
#if __has_builtin(__builtin_amdgcn_global_load_async_to_lds_b128) && \
    __has_builtin(__builtin_amdgcn_s_wait_asynccnt)
#define WL_ASYNC_LDS 1
#endif
#endif
#ifndef WL_ASYNC_LDS
#define WL_ASYNC_LDS 0
#endif

#if WL_ASYNC_LDS
typedef int wl_v4i __attribute__((vector_size(16)));
typedef __attribute__((address_space(1))) wl_v4i as1_v4i;
typedef __attribute__((address_space(3))) wl_v4i as3_v4i;
#define WL_AS1V4(p) ((as1_v4i*)(unsigned long long)(size_t)(p))
#define WL_AS3V4(p) ((as3_v4i*)(unsigned)(size_t)(p))
#endif

static __device__ __forceinline__ long long wl_mix(long long h, long long c1, long long c2) {
    h = (long long)((unsigned long long)h * (unsigned long long)c1);
    h ^= (h >> 31);
    h = (long long)((unsigned long long)h * (unsigned long long)c2);
    return h ^ (h >> 29);
}

static __device__ __forceinline__ int pack4b(int a, int b, int c, int d) {
    return (a & 0xFF) | ((b & 0xFF) << 8) | ((c & 0xFF) << 16) | ((d & 0xFF) << 24);
}

__global__ void wl_init_kernel(unsigned long long* __restrict__ s1,
                               unsigned long long* __restrict__ s2,
                               unsigned long long* __restrict__ keys,
                               unsigned* __restrict__ minidx,
                               int N, int T) {
    int i = blockIdx.x * blockDim.x + threadIdx.x;
    if (i < N) { s1[i] = 0ull; s2[i] = 0ull; }
    if (i < T) { keys[i] = KEY_EMPTY; minidx[i] = 0xFFFFFFFFu; }
}

__global__ void wl_label_kernel(const int* __restrict__ x,
                                long long* __restrict__ lab, int N) {
    const int gid  = blockIdx.x * blockDim.x + threadIdx.x;
    const int lane = threadIdx.x & 31;
    const int waveBase = gid - lane;

    unsigned long long W[8];
    {
        unsigned long long p = 1ull;
        #pragma unroll
        for (int j = 7; j >= 0; --j) { W[j] = p; p *= 1000003ull; }
    }

    v8i a = {0, 0, 0, 0, 0, 0, 0, 0};
    v8i b = {0, 0, 0, 0, 0, 0, 0, 0};
    v8i c = {0, 0, 0, 0, 0, 0, 0, 0};

    int n = (lane < 16) ? (lane & 7) : -1;
    if (n >= 0) {
        const int sh = 8 * n;
        int a0 = 0, a1 = 0;
        #pragma unroll
        for (int j = 0; j < 4; ++j) a0 |= ((int)((W[j]     >> sh) & 0xFF)) << (8 * j);
        #pragma unroll
        for (int j = 0; j < 4; ++j) a1 |= ((int)((W[4 + j] >> sh) & 0xFF)) << (8 * j);
        if (lane < 8) { a[0] = a0; a[1] = a1; } else { a[2] = a0; a[3] = a1; }
    }

    if (lane < 16) {
        const int nA = waveBase + lane;
        const int nB = nA + 16;
        if (nA < N) {
            const int4* p4 = (const int4*)(x + (size_t)nA * 8);
            int4 f0 = p4[0], f1 = p4[1];
            b[0] = pack4b(f0.x, f0.y, f0.z, f0.w);
            b[1] = pack4b(f1.x, f1.y, f1.z, f1.w);
        }
        if (nB < N) {
            const int4* p4 = (const int4*)(x + (size_t)nB * 8);
            int4 g0 = p4[0], g1 = p4[1];
            b[2] = pack4b(g0.x, g0.y, g0.z, g0.w);
            b[3] = pack4b(g1.x, g1.y, g1.z, g1.w);
        }
    }

    v8i d = __builtin_amdgcn_wmma_i32_16x16x64_iu8(false, a, false, b, c, false, false);

    unsigned long long labp = 0ull;
    #pragma unroll
    for (int r = 0; r < 8; ++r)
        labp += ((unsigned long long)(unsigned)d[r]) << (8 * r);

    long long l = wl_mix((long long)labp, WL_C1, WL_C2);
    if (gid < N) lab[gid] = l;
}

static __device__ __forceinline__ void wl_edge_one(int s, int d,
                                                   const long long* __restrict__ lab,
                                                   unsigned long long* __restrict__ s1,
                                                   unsigned long long* __restrict__ s2) {
    long long nl = lab[s];
    atomicAdd(&s1[d], (unsigned long long)wl_mix(nl, WL_C1, WL_C2));
    atomicAdd(&s2[d], (unsigned long long)wl_mix(nl, WL_C3, WL_C4));
}

__global__ void __launch_bounds__(256)
wl_edge_kernel(const int* __restrict__ ei,
               const long long* __restrict__ lab,
               unsigned long long* __restrict__ s1,
               unsigned long long* __restrict__ s2,
               int E, int nFull, int strideWaves) {
    const int lane = threadIdx.x & 31;
    const int wid  = threadIdx.x >> 5;
    const int gw   = blockIdx.x * (blockDim.x >> 5) + wid;
    const int* sp = ei;
    const int* dp = ei + E;

#if WL_ASYNC_LDS
    __shared__ int sA[8][2][128];
    __shared__ int sB[8][2][128];
    int chunk = gw;
    int buf = 0;
    if (chunk < nFull) {
        const int base = chunk * 128 + lane * 4;
        __builtin_amdgcn_global_load_async_to_lds_b128(WL_AS1V4(sp + base), WL_AS3V4(&sA[wid][0][lane * 4]), 0, 0);
        __builtin_amdgcn_global_load_async_to_lds_b128(WL_AS1V4(dp + base), WL_AS3V4(&sB[wid][0][lane * 4]), 0, 0);
    }
    while (chunk < nFull) {
        const int next = chunk + strideWaves;
        if (next < nFull) {
            const int nb = next * 128 + lane * 4;
            __builtin_amdgcn_global_load_async_to_lds_b128(WL_AS1V4(sp + nb), WL_AS3V4(&sA[wid][buf ^ 1][lane * 4]), 0, 0);
            __builtin_amdgcn_global_load_async_to_lds_b128(WL_AS1V4(dp + nb), WL_AS3V4(&sB[wid][buf ^ 1][lane * 4]), 0, 0);
            __builtin_amdgcn_s_wait_asynccnt(2);
        } else {
            __builtin_amdgcn_s_wait_asynccnt(0);
        }
        asm volatile("" ::: "memory");
        int4 s4 = *(const int4*)&sA[wid][buf][lane * 4];
        int4 d4 = *(const int4*)&sB[wid][buf][lane * 4];
        wl_edge_one(s4.x, d4.x, lab, s1, s2);
        wl_edge_one(s4.y, d4.y, lab, s1, s2);
        wl_edge_one(s4.z, d4.z, lab, s1, s2);
        wl_edge_one(s4.w, d4.w, lab, s1, s2);
        buf ^= 1;
        chunk = next;
    }
#else
    for (int chunk = gw; chunk < nFull; chunk += strideWaves) {
        const int base = chunk * 128 + lane * 4;
        __builtin_prefetch(sp + base + 16384, 0, 0);
        __builtin_prefetch(dp + base + 16384, 0, 0);
        int4 s4 = *(const int4*)(sp + base);
        int4 d4 = *(const int4*)(dp + base);
        wl_edge_one(s4.x, d4.x, lab, s1, s2);
        wl_edge_one(s4.y, d4.y, lab, s1, s2);
        wl_edge_one(s4.z, d4.z, lab, s1, s2);
        wl_edge_one(s4.w, d4.w, lab, s1, s2);
    }
#endif

    if (gw == 0) {
        for (int k = nFull * 128 + lane; k < E; k += 32)
            wl_edge_one(sp[k], dp[k], lab, s1, s2);
    }
}

__global__ void wl_insert_kernel(const long long* __restrict__ lab,
                                 const unsigned long long* __restrict__ s1,
                                 const unsigned long long* __restrict__ s2,
                                 unsigned long long* __restrict__ keys,
                                 unsigned* __restrict__ minidx,
                                 unsigned* __restrict__ slotof, int N) {
    int i = blockIdx.x * blockDim.x + threadIdx.x;
    if (i >= N) return;
    long long l = lab[i];
    long long t1 = (long long)((unsigned long long)l * (unsigned long long)WL_C3 + s1[i]);
    long long t2 = (long long)(s2[i] + (unsigned long long)l);
    long long h = wl_mix(t1, WL_C1, WL_C2) ^ wl_mix(t2, WL_C3, WL_C4);

    unsigned long long key = (unsigned long long)h;
    unsigned slot = (unsigned)((key * 0x9E3779B97F4A7C15ull) >> (64 - TAB_BITS));
    for (;;) {
        unsigned long long prev = atomicCAS(&keys[slot], KEY_EMPTY, key);
        if (prev == KEY_EMPTY || prev == key) break;
        slot = (slot + 1u) & (TAB_SIZE - 1u);
    }
    atomicMin(&minidx[slot], (unsigned)i);
    slotof[i] = slot;
}

__global__ void wl_flag_kernel(const unsigned* __restrict__ slotof,
                               const unsigned* __restrict__ minidx,
                               unsigned* __restrict__ pfx, int N) {
    int i = blockIdx.x * blockDim.x + threadIdx.x;
    if (i >= N) return;
    pfx[i] = (minidx[slotof[i]] == (unsigned)i) ? 1u : 0u;
}

__global__ void wl_scan_blocks(unsigned* __restrict__ data,
                               unsigned* __restrict__ sums, int N) {
    __shared__ unsigned sh[1024];
    const int tid = threadIdx.x;
    const int i = blockIdx.x * 1024 + tid;
    unsigned v = (i < N) ? data[i] : 0u;
    sh[tid] = v;
    __syncthreads();
    for (int off = 1; off < 1024; off <<= 1) {
        unsigned t = (tid >= off) ? sh[tid - off] : 0u;
        __syncthreads();
        sh[tid] += t;
        __syncthreads();
    }
    unsigned incl = sh[tid];
    if (i < N) data[i] = incl - v;
    if (tid == 1023) sums[blockIdx.x] = incl;
}

__global__ void wl_scan_sums(unsigned* __restrict__ sums, int nb) {
    __shared__ unsigned sh[1024];
    const int tid = threadIdx.x;
    unsigned v = (tid < nb) ? sums[tid] : 0u;
    sh[tid] = v;
    __syncthreads();
    for (int off = 1; off < 1024; off <<= 1) {
        unsigned t = (tid >= off) ? sh[tid - off] : 0u;
        __syncthreads();
        sh[tid] += t;
        __syncthreads();
    }
    if (tid < nb) sums[tid] = sh[tid] - v;
}

__global__ void wl_scan_add(unsigned* __restrict__ data,
                            const unsigned* __restrict__ sums, int N) {
    int i = blockIdx.x * 1024 + threadIdx.x;
    if (i < N) data[i] += sums[blockIdx.x];
}

__global__ void wl_output_kernel(const unsigned* __restrict__ slotof,
                                 const unsigned* __restrict__ minidx,
                                 const unsigned* __restrict__ pfx,
                                 float* __restrict__ out, int N) {
    int i = blockIdx.x * blockDim.x + threadIdx.x;
    if (i >= N) return;
    unsigned rep = minidx[slotof[i]];
    out[i] = (float)pfx[rep];
}

extern "C" void kernel_launch(void* const* d_in, const int* in_sizes, int n_in,
                              void* d_out, int out_size, void* d_ws, size_t ws_size,
                              hipStream_t stream) {
    const int* x  = (const int*)d_in[0];
    const int* ei = (const int*)d_in[1];
    const int N = in_sizes[0] / 8;
    const int E = in_sizes[1] / 2;
    float* out = (float*)d_out;

    char* ws = (char*)d_ws;
    long long*           lab    = (long long*)(ws);
    unsigned long long*  s1     = (unsigned long long*)(ws + 8ull * N);
    unsigned long long*  s2     = (unsigned long long*)(ws + 16ull * N);
    unsigned*            slotof = (unsigned*)(ws + 24ull * N);
    unsigned long long*  keys   = (unsigned long long*)(ws + 28ull * N);
    unsigned*            minidx = (unsigned*)(ws + 28ull * N + 8ull * TAB_SIZE);
    unsigned*            pfx    = (unsigned*)(ws + 28ull * N + 12ull * TAB_SIZE);
    unsigned*            sums   = (unsigned*)(ws + 32ull * N + 12ull * TAB_SIZE);

    const int B = 256;
    const int initM = (N > (int)TAB_SIZE) ? N : (int)TAB_SIZE;

    wl_init_kernel<<<(initM + B - 1) / B, B, 0, stream>>>(s1, s2, keys, minidx, N, (int)TAB_SIZE);

    wl_label_kernel<<<(N + B - 1) / B, B, 0, stream>>>(x, lab, N);

    const int nFull = ((E & 3) == 0) ? (E >> 7) : 0;
    const int eBlocks = 1024;
    wl_edge_kernel<<<eBlocks, B, 0, stream>>>(ei, lab, s1, s2, E, nFull, eBlocks * (B / 32));

    wl_insert_kernel<<<(N + B - 1) / B, B, 0, stream>>>(lab, s1, s2, keys, minidx, slotof, N);

    wl_flag_kernel<<<(N + B - 1) / B, B, 0, stream>>>(slotof, minidx, pfx, N);

    const int nb = (N + 1023) / 1024;
    wl_scan_blocks<<<nb, 1024, 0, stream>>>(pfx, sums, N);
    wl_scan_sums<<<1, 1024, 0, stream>>>(sums, nb);
    wl_scan_add<<<nb, 1024, 0, stream>>>(pfx, sums, N);

    wl_output_kernel<<<(N + B - 1) / B, B, 0, stream>>>(slotof, minidx, pfx, out, N);
}
